// GNNModel_24919400251471
// MI455X (gfx1250) — matrix operand order repaired and checked by reading, not compiled, not run on hardware
//
#include <hip/hip_runtime.h>
#include <hip/hip_bf16.h>

// ---------------------------------------------------------------------------
// GNN (2x GAT + pool + MLP) for MI455X / gfx1250, wave32, WMMA f16->f32,
// double-buffered GLOBAL_LOAD_ASYNC_TO_LDS for the Wh tiles.
// B=16, N=1024, F_IN=7, EMB=256, HEADS=8, DIM=32, POOL=512.
// ---------------------------------------------------------------------------

typedef __attribute__((ext_vector_type(16))) _Float16 v16h;
typedef __attribute__((ext_vector_type(8)))  _Float16 h8;
typedef __attribute__((ext_vector_type(8)))  float    v8f;
typedef __attribute__((ext_vector_type(4)))  float    f4;

#define NB   16
#define NN   1024
#define NH   8
#define ND   32
#define EMB  256

// ---------------- workspace layout (bytes) ----------------
#define OFF_WHT1   (size_t)0                      // f16 [B*256][1024]   8 MB
#define OFF_WHT2   (OFF_WHT1 + 8388608)           // f16 [B*256][1024]   8 MB
#define OFF_H1     (OFF_WHT2 + 8388608)           // f16 [B*N][256]      8 MB
#define OFF_H2     (OFF_H1   + 8388608)           // f16 [B*N][256]      8 MB
#define OFF_S1     (OFF_H2   + 8388608)           // f32 [B*H][N]      512 KB
#define OFF_S2     (OFF_S1   + 524288)            // f32 [B*H][N]      512 KB
#define OFF_AMAX   (OFF_S2   + 524288)            // f32 [B*N]          64 KB
#define OFF_AMIN   (OFF_AMAX + 65536)             // f32 [B*N]          64 KB
#define OFF_W2T    (OFF_AMIN + 65536)             // f16 [256][256]    128 KB
#define OFF_P      (OFF_W2T  + 131072)            // f32 [16][256]      16 KB

// ---------------- adj row max/min (one wave per row) ----------------
__global__ __launch_bounds__(256) void adjstat_kernel(
    const float* __restrict__ adj, float* __restrict__ amax, float* __restrict__ amin)
{
    int wave = threadIdx.x >> 5, lane = threadIdx.x & 31;
    int rid  = blockIdx.x * 8 + wave;                 // 0 .. B*N-1
    const f4* row = (const f4*)(adj + (size_t)rid * NN);
    float mx = -1e30f, mn = 1e30f;
#pragma unroll
    for (int k = 0; k < 8; ++k) {
        f4 v = row[k * 32 + lane];
#pragma unroll
        for (int c = 0; c < 4; ++c) { mx = fmaxf(mx, v[c]); mn = fminf(mn, v[c]); }
    }
#pragma unroll
    for (int off = 16; off; off >>= 1) {
        mx = fmaxf(mx, __shfl_xor(mx, off, 32));
        mn = fminf(mn, __shfl_xor(mn, off, 32));
    }
    if (lane == 0) { amax[rid] = mx; amin[rid] = mn; }
}

// ---------------- layer-1 projection (K=7, scalar f32), transposed f16 out --
__global__ __launch_bounds__(256) void proj1_kernel(
    const float* __restrict__ x,    // [B,N,7]
    const float* __restrict__ Ws,   // [H,7,32]
    const float* __restrict__ bsv,  // [H,32] flat
    _Float16* __restrict__ whT)     // [B*256][1024]
{
    int bn = blockIdx.x;                 // b*1024+n
    int t  = threadIdx.x;                // h*32+d
    int h  = t >> 5, d = t & 31;
    const float* xp = x + (size_t)bn * 7;
    float acc = bsv[t];
#pragma unroll
    for (int i = 0; i < 7; ++i) acc += xp[i] * Ws[h * 224 + i * 32 + d];
    int b = bn >> 10, n = bn & 1023;
    whT[((size_t)(b * 256 + t)) * NN + n] = (_Float16)acc;
}

// ---------------- s[b,h,n] = Wh . (q_lo+q_hi) + qb ----------------
__global__ __launch_bounds__(256) void s_kernel(
    const _Float16* __restrict__ whT,  // [B*H*32][1024]
    const float* __restrict__ q,       // [H,64]
    const float* __restrict__ qb,      // [H]
    float* __restrict__ s)             // [B*H][1024]
{
    int t  = blockIdx.x * 256 + threadIdx.x;  // bh*1024 + n
    int bh = t >> 10, n = t & 1023;
    int h  = bh & 7;
    float acc = qb[h];
    const _Float16* wp = whT + (size_t)bh * 32 * NN + n;
#pragma unroll
    for (int d = 0; d < ND; ++d)
        acc += (float)wp[(size_t)d * NN] * (q[h * 64 + d] + q[h * 64 + 32 + d]);
    s[t] = acc;
}

// ---------------- flash-style GAT attention aggregation ----------------
// 4 waves / block; wave w owns rows i0 = (bx*4+w)*16 of one (b,h).
// Waves share the Wh j-chunk (32x32 f16) via double-buffered async-to-LDS.
__global__ __launch_bounds__(128) void gat_attn_kernel(
    const float* __restrict__ adj,      // [B,N,N]
    const float* __restrict__ s,        // [B*H][N]
    const float* __restrict__ amax,     // [B*N]
    const float* __restrict__ amin,     // [B*N]
    const _Float16* __restrict__ whT,   // [B*H*32][N]  (row = d, col = j)
    _Float16* __restrict__ hout)        // [B,N,256]
{
    __shared__ _Float16 ldsB[2][32 * 32];  // double buffer, [d][j]
    const int wave = threadIdx.x >> 5;
    const int lane = threadIdx.x & 31;
    const int h = blockIdx.y, b = blockIdx.z;
    const int i0 = (blockIdx.x * 4 + wave) * 16;
    const int r  = lane & 15;
    const bool hi = lane >= 16;
    const int aoff = hi ? 8 : 0;
    const int kb   = hi ? 8 : 0;
    const int bh   = b * NH + h;

    // per-row scalars (row = i0+r for this lane)
    float sr = s[bh * NN + i0 + r];
    float zx = sr * (sr >= 0.f ? amax[b * NN + i0 + r] : amin[b * NN + i0 + r]);
    float mrow = zx > 0.f ? zx : 0.01f * zx;     // row max of leaky_relu(adj*s)

    const float*    arow  = adj + ((size_t)(b * NN + i0 + r)) * NN;
    const _Float16* wbase = whT + ((size_t)bh * 32) * NN;

    // this thread's piece of the cooperative 32x32 f16 tile (8 halves = 16 B)
    const int td = threadIdx.x >> 2;           // d row 0..31
    const int tj = (threadIdx.x & 3) * 8;      // j offset 0,8,16,24
    const unsigned goff  = (unsigned)((td * NN + tj) * 2);   // bytes into wbase+jc
    // LDS byte offsets (flat-LDS aperture: addr[31:0] == wave-relative offset)
    const unsigned loff0 = (unsigned)(size_t)(&ldsB[0][td * 32 + tj]);
    const unsigned loff1 = (unsigned)(size_t)(&ldsB[1][td * 32 + tj]);

#define ISSUE_ASYNC(LOFF, JC)                                                    \
    {                                                                            \
        unsigned long long sbase_ = (unsigned long long)(wbase + (JC));          \
        asm volatile("global_load_async_to_lds_b128 %0, %1, %2"                  \
                     :: "v"(LOFF), "v"(goff), "s"(sbase_) : "memory");           \
    }

    v8f acc0 = {}; v8f acc1 = {};
    float se0 = 0.f, se1 = 0.f, se2 = 0.f, se3 = 0.f;

    // prologue: fill buffer 0 with chunk 0
    ISSUE_ASYNC(loff0, 0)
    asm volatile("s_wait_asynccnt 0x0" ::: "memory");
    __syncthreads();

    int cur = 0;
    for (int jc = 0; jc < NN; jc += 32) {
        // kick off the next tile into the other buffer (overlaps with compute)
        if (jc + 32 < NN) ISSUE_ASYNC(cur ? loff0 : loff1, jc + 32)
        if (jc + 64 < NN) __builtin_prefetch(arow + jc + 64, 0, 3);

        // A fragment: p = exp(lrelu(adj*s) - mrow), packed to f16
        float av[16] __attribute__((aligned(16)));
        *(f4*)(&av[0])  = *(const f4*)(arow + jc + aoff);
        *(f4*)(&av[4])  = *(const f4*)(arow + jc + aoff + 4);
        *(f4*)(&av[8])  = *(const f4*)(arow + jc + aoff + 16);
        *(f4*)(&av[12]) = *(const f4*)(arow + jc + aoff + 20);
        v16h afrag;
#pragma unroll
        for (int k = 0; k < 16; k += 4) {
            float z0 = av[k] * sr,     z1 = av[k + 1] * sr;
            float z2 = av[k + 2] * sr, z3 = av[k + 3] * sr;
            float p0 = __expf((z0 > 0.f ? z0 : 0.01f * z0) - mrow);
            float p1 = __expf((z1 > 0.f ? z1 : 0.01f * z1) - mrow);
            float p2 = __expf((z2 > 0.f ? z2 : 0.01f * z2) - mrow);
            float p3 = __expf((z3 > 0.f ? z3 : 0.01f * z3) - mrow);
            se0 += p0; se1 += p1; se2 += p2; se3 += p3;
            afrag[k]     = (_Float16)p0;
            afrag[k + 1] = (_Float16)p1;
            afrag[k + 2] = (_Float16)p2;
            afrag[k + 3] = (_Float16)p3;
        }

        // B fragments from current LDS buffer: col N = r (+16), K = kb+0..7, kb+16..23
        const _Float16* lb = &ldsB[cur][0];
        v16h bf0, bf1;
        const _Float16* pb0 = lb + r * 32 + kb;
        const _Float16* pb1 = lb + (16 + r) * 32 + kb;
#pragma unroll
        for (int i = 0; i < 16; ++i) { bf0[i] = pb0[i < 8 ? i : i + 8]; bf1[i] = pb1[i < 8 ? i : i + 8]; }

        acc0 = __builtin_amdgcn_wmma_f32_16x16x32_f16(false, afrag, false, bf0,
                                                      (short)0, acc0, false, false);
        acc1 = __builtin_amdgcn_wmma_f32_16x16x32_f16(false, afrag, false, bf1,
                                                      (short)0, acc1, false, false);

        // next buffer ready + everyone done with this one
        asm volatile("s_wait_asynccnt 0x0" ::: "memory");
        __syncthreads();
        cur ^= 1;
    }
#undef ISSUE_ASYNC

    // softmax denominator: combine half-rows, one reciprocal per lane
    float separt = (se0 + se1) + (se2 + se3);
    float se   = separt + __shfl_xor(separt, 16, 32);
    float rinv = 1.0f / se;            // valid for row r in lanes r and r+16

    // C layout: lane<16 -> (M=v, N=lane); lane>=16 -> (M=v+8, N=lane-16)
#pragma unroll
    for (int v = 0; v < 8; ++v) {
        int M = v + (hi ? 8 : 0);
        float inv = __shfl(rinv, M, 32);
        size_t rowb = (size_t)(b * NN + i0 + M) * EMB;
        hout[rowb + h * 32 + r]      = (_Float16)(acc0[v] * inv);
        hout[rowb + h * 32 + 16 + r] = (_Float16)(acc1[v] * inv);
    }
}

// ---------------- transpose/convert layer-2 weights ----------------
__global__ __launch_bounds__(256) void prepw2_kernel(
    const float* __restrict__ Wm, _Float16* __restrict__ w2t)  // w2t[e][k]
{
    int t = blockIdx.x * 256 + threadIdx.x;   // e*256 + k
    int e = t >> 8, k = t & 255;
    w2t[t] = (_Float16)Wm[(e >> 5) * 8192 + k * 32 + (e & 31)];
}

// ---------------- layer-2 projection: WMMA GEMM [16384,256]x[256,256] -------
__global__ __launch_bounds__(128) void proj2_kernel(
    const _Float16* __restrict__ h1,   // [B*N][256] row-major
    const _Float16* __restrict__ w2t,  // [256][256]  (e-major)
    const float* __restrict__ bmf,     // [256]
    _Float16* __restrict__ whT2)       // [B*256][1024]
{
    int wave = threadIdx.x >> 5, lane = threadIdx.x & 31;
    int m0 = (blockIdx.x * 4 + wave) * 16;
    int c0 = blockIdx.y * 16;
    int r = lane & 15;
    bool hi = lane >= 16;
    const int e = c0 + r;
    const int aoff = hi ? 8 : 0;
    const int boff = hi ? 8 : 0;

    const _Float16* arow = h1  + (size_t)(m0 + r) * 256;
    const _Float16* brow = w2t + (size_t)e * 256;

    v8f acc = {};
    for (int kc = 0; kc < 256; kc += 32) {
        h8 alo = *(const h8*)(arow + kc + aoff);
        h8 ahi = *(const h8*)(arow + kc + aoff + 16);
        h8 blo = *(const h8*)(brow + kc + boff);
        h8 bhi = *(const h8*)(brow + kc + boff + 16);
        v16h a, bf;
#pragma unroll
        for (int i = 0; i < 8; ++i) {
            a[i] = alo[i]; a[8 + i] = ahi[i];
            bf[i] = blo[i]; bf[8 + i] = bhi[i];
        }
        acc = __builtin_amdgcn_wmma_f32_16x16x32_f16(false, a, false, bf,
                                                     (short)0, acc, false, false);
    }
    float bias = bmf[e];
    int b    = m0 >> 10;
    int nloc = (m0 & 1023) + (hi ? 8 : 0);
    h8 st;
#pragma unroll
    for (int v = 0; v < 8; ++v) st[v] = (_Float16)(acc[v] + bias);
    *(h8*)(&whT2[((size_t)(b * 256 + e)) * NN + nloc]) = st;
}

// ---------------- sum-pool over nodes ----------------
__global__ __launch_bounds__(256) void pool_kernel(
    const _Float16* __restrict__ h2, float* __restrict__ p)
{
    int b = blockIdx.x, e = threadIdx.x;
    const _Float16* hp = h2 + (size_t)b * NN * EMB + e;
    float acc = 0.f;
    for (int n = 0; n < NN; ++n) acc += (float)hp[(size_t)n * EMB];
    p[b * EMB + e] = acc;
}

// ---------------- MLP head (tiny; single block, f32 with LDS staging) -------
__global__ __launch_bounds__(512) void mlp_kernel(
    const float* __restrict__ p,
    const float* __restrict__ Wp, const float* __restrict__ bp,
    const float* __restrict__ gamma, const float* __restrict__ beta,
    const float* __restrict__ rmean, const float* __restrict__ rvar,
    const float* __restrict__ W1, const float* __restrict__ b1,
    const float* __restrict__ W2, const float* __restrict__ b2,
    float* __restrict__ out)
{
    __shared__ float A[4096];    // p [16,256] then y1 [16,256]
    __shared__ float Bf[8192];   // y0 [16,512]
    __shared__ float C2[2048];   // y2 [16,128]
    int tid = threadIdx.x;

    for (int i = tid; i < 4096; i += 512) A[i] = p[i];
    __syncthreads();

    // y0 = relu(BN(p @ Wp + bp))  [16,512]
    for (int idx = tid; idx < 8192; idx += 512) {
        int rr = idx >> 9, c = idx & 511;
        float acc = bp[c];
        for (int k = 0; k < 256; ++k) acc += A[rr * 256 + k] * Wp[k * 512 + c];
        acc = (acc - rmean[c]) * rsqrtf(rvar[c] + 1e-5f) * gamma[c] + beta[c];
        Bf[idx] = fmaxf(acc, 0.f);
    }
    __syncthreads();

    // y1 = relu(y0 @ W1 + b1)  [16,256]
    for (int idx = tid; idx < 4096; idx += 512) {
        int rr = idx >> 8, c = idx & 255;
        float acc = b1[c];
        for (int k = 0; k < 512; ++k) acc += Bf[rr * 512 + k] * W1[k * 256 + c];
        A[idx] = fmaxf(acc, 0.f);
    }
    __syncthreads();

    // y2 = relu(y1 @ W2 + b2)  [16,128]
    for (int idx = tid; idx < 2048; idx += 512) {
        int rr = idx >> 7, c = idx & 127;
        float acc = b2[c];
        for (int k = 0; k < 256; ++k) acc += A[rr * 256 + k] * W2[k * 128 + c];
        C2[idx] = fmaxf(acc, 0.f);
    }
    __syncthreads();

    if (tid < NB) {
        float acc = 0.f;
        for (int k = 0; k < 128; ++k) acc += C2[tid * 128 + k];
        out[tid] = acc * (1.0f / 128.0f);
    }
}

// ---------------------------------------------------------------------------
extern "C" void kernel_launch(void* const* d_in, const int* in_sizes, int n_in,
                              void* d_out, int out_size, void* d_ws, size_t ws_size,
                              hipStream_t stream) {
    (void)in_sizes; (void)n_in; (void)out_size; (void)ws_size;
    const float* x     = (const float*)d_in[0];
    const float* adj   = (const float*)d_in[1];
    const float* Ws    = (const float*)d_in[2];
    const float* bsv   = (const float*)d_in[3];
    const float* qs    = (const float*)d_in[4];
    const float* qbs   = (const float*)d_in[5];
    const float* Wm    = (const float*)d_in[6];
    const float* bm    = (const float*)d_in[7];
    const float* qm    = (const float*)d_in[8];
    const float* qbm   = (const float*)d_in[9];
    const float* Wp    = (const float*)d_in[10];
    const float* bp    = (const float*)d_in[11];
    const float* gamma = (const float*)d_in[12];
    const float* beta  = (const float*)d_in[13];
    const float* rmean = (const float*)d_in[14];
    const float* rvar  = (const float*)d_in[15];
    const float* W1    = (const float*)d_in[16];
    const float* b1    = (const float*)d_in[17];
    const float* W2    = (const float*)d_in[18];
    const float* b2    = (const float*)d_in[19];
    float* out = (float*)d_out;

    char* ws = (char*)d_ws;
    _Float16* whT1 = (_Float16*)(ws + OFF_WHT1);
    _Float16* whT2 = (_Float16*)(ws + OFF_WHT2);
    _Float16* h1   = (_Float16*)(ws + OFF_H1);
    _Float16* h2   = (_Float16*)(ws + OFF_H2);
    float*    s1   = (float*)(ws + OFF_S1);
    float*    s2   = (float*)(ws + OFF_S2);
    float*    am   = (float*)(ws + OFF_AMAX);
    float*    an   = (float*)(ws + OFF_AMIN);
    _Float16* w2t  = (_Float16*)(ws + OFF_W2T);
    float*    pbuf = (float*)(ws + OFF_P);

    adjstat_kernel<<<2048, 256, 0, stream>>>(adj, am, an);
    prepw2_kernel<<<256, 256, 0, stream>>>(Wm, w2t);
    proj1_kernel<<<NB * NN, 256, 0, stream>>>(x, Ws, bsv, whT1);
    s_kernel<<<512, 256, 0, stream>>>(whT1, qs, qbs, s1);
    gat_attn_kernel<<<dim3(16, NH, NB), 128, 0, stream>>>(adj, s1, am, an, whT1, h1);
    proj2_kernel<<<dim3(256, 16), 128, 0, stream>>>(h1, w2t, bm, whT2);
    s_kernel<<<512, 256, 0, stream>>>(whT2, qm, qbm, s2);
    gat_attn_kernel<<<dim3(16, NH, NB), 128, 0, stream>>>(adj, s2, am, an, whT2, h2);
    pool_kernel<<<NB, 256, 0, stream>>>(h2, pbuf);
    mlp_kernel<<<1, 512, 0, stream>>>(pbuf, Wp, bp, gamma, beta, rmean, rvar,
                                      W1, b1, W2, b2, out);
}
